// MAlphaAttention_50173807952661
// MI455X (gfx1250) — matrix operand order repaired and checked by reading, not compiled, not run on hardware
//
#include <hip/hip_runtime.h>

// ---------------------------------------------------------------------------
// MI455X (gfx1250) implementation of MAlphaAttention.
// - All GEMMs: v_wmma_f32_16x16x32_bf16 (bf16 operands, f32 accumulation).
// - Static operands pre-transposed/converted once; Q/K/V written transposed,
//   so EVERY GEMM tile fill (A and B) is GLOBAL_LOAD_ASYNC_TO_LDS_B128.
// - GEMM + flash K-loops double-buffered in LDS (one barrier per step).
// - [B,H,N,N] masked kernel never materialized (flash-style fusion).
// ---------------------------------------------------------------------------

typedef unsigned short u16;
typedef unsigned int u32;
typedef __attribute__((ext_vector_type(16))) __bf16 bf16x16;
typedef __attribute__((ext_vector_type(8))) float f32x8;
typedef __attribute__((ext_vector_type(4))) u32 u32x4;

union FragB16 { bf16x16 v; u32x4 q[2]; };
union U16x8 { u16 h[8]; u32x4 v; };

#define NPATCH 1024
#define DIMC 768
#define BN 4096            // B * N
#define SEG (4096 * 768)   // elements in one [B*N, C] (or transposed) buffer

__device__ __forceinline__ u16 f2bf(float f) {            // hw v_cvt path
  return __builtin_bit_cast(u16, (__bf16)f);
}
__device__ __forceinline__ float bf2f(u16 h) {
  u32 u = ((u32)h) << 16;
  return __builtin_bit_cast(float, u);
}
__device__ __forceinline__ f32x8 fz8() {
  return f32x8{0.f, 0.f, 0.f, 0.f, 0.f, 0.f, 0.f, 0.f};
}
__device__ __forceinline__ f32x8 wmma_bf16(const FragB16& a, const FragB16& b, f32x8 c) {
  return __builtin_amdgcn_wmma_f32_16x16x32_bf16(false, a.v, false, b.v,
                                                 (short)0, c, false, false);
}

// Async global->LDS 128b copy (gfx1250, ASYNCcnt-tracked). LDS byte address =
// low 32 bits of the generic pointer (ISA 10.2: LDS aperture addr = addr[31:0]).
__device__ __forceinline__ void async_copy_b128(const u16* gsrc, const u16* ldst) {
  u32 lds = (u32)(size_t)(const void*)ldst;
  asm volatile("global_load_async_to_lds_b128 %0, %1, off"
               :
               : "v"(lds), "v"(gsrc)
               : "memory");
}
__device__ __forceinline__ void wait_async0() {
  asm volatile("s_wait_asynccnt 0" ::: "memory");
}

// ---------------------------------------------------------------------------
// f32 -> bf16 bulk convert (4 elems/thread, hw cvt) -- for x only
// ---------------------------------------------------------------------------
__global__ __launch_bounds__(256) void cvt_f32_bf16(const float* __restrict__ in,
                                                    u16* __restrict__ out, int n) {
  int i = (blockIdx.x * 256 + threadIdx.x) * 4;
  if (i < n) {
    const float4 v = *(const float4*)(in + i);
    u16 o[4] = {f2bf(v.x), f2bf(v.y), f2bf(v.z), f2bf(v.w)};
    *(uint2*)(out + i) = *(const uint2*)o;
  }
}

// ---------------------------------------------------------------------------
// Transpose + convert: in [R][C] f32 -> out [C][R] bf16.
// Used for mask^T, Wqkv^T, Wout^T (one-time, static operands).
// ---------------------------------------------------------------------------
__global__ __launch_bounds__(256) void transpose_cvt(const float* __restrict__ in,
                                                     u16* __restrict__ out,
                                                     int R, int C) {
  __shared__ float tile[32][33];
  const int bx = blockIdx.x * 32;  // C dim base
  const int by = blockIdx.y * 32;  // R dim base
  const int tx = threadIdx.x & 31;
  const int ty = threadIdx.x >> 5;  // 0..7
#pragma unroll
  for (int j = 0; j < 4; ++j)
    tile[ty + 8 * j][tx] = in[(size_t)(by + ty + 8 * j) * C + bx + tx];
  __syncthreads();
#pragma unroll
  for (int j = 0; j < 4; ++j)
    out[(size_t)(bx + ty + 8 * j) * R + by + tx] = f2bf(tile[tx][ty + 8 * j]);
}

// ---------------------------------------------------------------------------
// 128x128 WMMA GEMM, all-bf16, 8 waves, wave tile 64x32, BK=32.
// Both A and B(^T) are row-copied into LDS with async-to-LDS b128;
// LDS double-buffered, one barrier per K-step.
// MODE 0: qkv; out written TRANSPOSED into QT/KT/VT segs (relu+eps on q,k),
//         vectorized b128 stores (8 consecutive rows per lane).
// MODE 1: graph mix; out(row-major bf16) = srcT + 0.1*acc, src read
//         vectorized from the transposed B buffer; grid.z = b*2 + {q,k}.
// MODE 2: out(f32, row-major) = acc + bias.
// ---------------------------------------------------------------------------
template <int MODE>
__global__ __launch_bounds__(256) void wgemm128(const u16* __restrict__ Ag,
                                                const u16* __restrict__ Bg,
                                                void* __restrict__ Og_,
                                                const void* __restrict__ Xg_,
                                                int lda, int ldbt, int K) {
  constexpr int LT = 40;  // 32 + 8 pad (80B row: 16B aligned, conflict-free b128)
  __shared__ u16 At[2][128 * LT];  // [row][k]
  __shared__ u16 Bt[2][128 * LT];  // [col][k]  (source already transposed)

  const int t = threadIdx.x;
  const int lane = t & 31, wave = t >> 5;
  const int l16 = lane & 15, half = lane >> 4;
  const int wm = wave >> 2, wn = wave & 3;
  const int n0 = blockIdx.x * 128;
  const int m0 = blockIdx.y * 128;

  size_t boffB = 0, boffO = 0;
  if constexpr (MODE == 1) {
    const int z = blockIdx.z;
    boffB = (size_t)(z & 1) * SEG + (size_t)(z >> 1) * NPATCH;          // QT/KT col slice
    boffO = (size_t)(z & 1) * SEG + (size_t)(z >> 1) * (NPATCH * DIMC); // Q'/K' row slice
  }
  const u16* Abase = Ag;
  const u16* Bbase = Bg + boffB;

  f32x8 acc[4][2];
#pragma unroll
  for (int i = 0; i < 4; ++i)
#pragma unroll
    for (int j = 0; j < 2; ++j) acc[i][j] = fz8();

  auto fill = [&](int kk, int buf) {
    // A tile: 128 rows x 32 k = 8KB, 2 async b128 per thread
#pragma unroll
    for (int i = 0; i < 2; ++i) {
      int idx = t + i * 256;
      int row = idx >> 2, kc = (idx & 3) * 8;
      async_copy_b128(Abase + (size_t)(m0 + row) * lda + kk + kc,
                      &At[buf][row * LT + kc]);
    }
    // B tile (from transposed operand): 128 cols x 32 k, 2 async b128 per thread
#pragma unroll
    for (int i = 0; i < 2; ++i) {
      int idx = t + i * 256;
      int row = idx >> 2, kc = (idx & 3) * 8;
      async_copy_b128(Bbase + (size_t)(n0 + row) * ldbt + kk + kc,
                      &Bt[buf][row * LT + kc]);
    }
  };

  fill(0, 0);
  const int nk = K >> 5;
  for (int kt = 0; kt < nk; ++kt) {
    const int cur = kt & 1;
    wait_async0();        // this wave's async fills (invisible to compiler)
    __syncthreads();      // all waves' fills done
    if (kt + 1 < nk) fill((kt + 1) << 5, cur ^ 1);

    // A frag: lane row = l16; K = 8*half..+7 (VGPR0-3), 16+8*half..+7 (VGPR4-7)
    FragB16 afr[4], bfr[2];
#pragma unroll
    for (int mt = 0; mt < 4; ++mt) {
      const u16* p = &At[cur][(wm * 64 + mt * 16 + l16) * LT + 8 * half];
      afr[mt].q[0] = *(const u32x4*)p;
      afr[mt].q[1] = *(const u32x4*)(p + 16);
    }
    // B frag: lane col = l16; K = 8*half..+7 (VGPR0-3), 16+8*half..+7 (VGPR4-7)
#pragma unroll
    for (int nt = 0; nt < 2; ++nt) {
      const u16* p = &Bt[cur][(wn * 32 + nt * 16 + l16) * LT + 8 * half];
      bfr[nt].q[0] = *(const u32x4*)p;
      bfr[nt].q[1] = *(const u32x4*)(p + 16);
    }
#pragma unroll
    for (int mt = 0; mt < 4; ++mt)
#pragma unroll
      for (int nt = 0; nt < 2; ++nt)
        acc[mt][nt] = wmma_bf16(afr[mt], bfr[nt], acc[mt][nt]);
  }

  // ---- epilogue (C layout: VGPR r, lane -> row rb+r, col l16) ----
#pragma unroll
  for (int mt = 0; mt < 4; ++mt) {
#pragma unroll
    for (int nt = 0; nt < 2; ++nt) {
      const int col = n0 + wn * 32 + nt * 16 + l16;
      const int rb = m0 + wm * 64 + mt * 16 + 8 * half;  // multiple of 8
      if constexpr (MODE == 0) {
        // transposed write: 8 consecutive rows at fixed col -> one b128
        const int seg = col / DIMC, nn = col - seg * DIMC;
        U16x8 pk;
#pragma unroll
        for (int r = 0; r < 8; ++r) {
          float v = acc[mt][nt][r];
          if (seg < 2) v = fmaxf(v, 0.f) + 1e-6f;  // relu+eps on q,k
          pk.h[r] = f2bf(v);
        }
        *(u32x4*)((u16*)Og_ + (size_t)seg * SEG + (size_t)nn * BN + rb) = pk.v;
      } else if constexpr (MODE == 1) {
        // src q value: transposed buffer, 8 consecutive rows -> one b128 read
        U16x8 raw;
        raw.v = *(const u32x4*)(Bbase + (size_t)col * BN + rb);
        u16* op = (u16*)Og_ + boffO;
#pragma unroll
        for (int r = 0; r < 8; ++r)
          op[(size_t)(rb + r) * DIMC + col] =
              f2bf(bf2f(raw.h[r]) + 0.1f * acc[mt][nt][r]);
      } else {
#pragma unroll
        for (int r = 0; r < 8; ++r)
          ((float*)Og_)[(size_t)(rb + r) * DIMC + col] =
              acc[mt][nt][r] + ((const float*)Xg_)[col];
      }
    }
  }
}

// ---------------------------------------------------------------------------
// Fused masked linear attention (flash-style, no N x N intermediate).
// Block = one (b,h) x 128 q-rows; 8 waves, each owns a 16-row strip.
// K' chunk (row-major) and V chunk (from transposed VT) both staged with
// async-to-LDS b128, double-buffered. Per chunk: S = q@k^T (WMMA), S *= mask,
// rowsum += S, O += S@V (WMMA) with S relaid out via per-wave LDS scratch.
// ---------------------------------------------------------------------------
__global__ __launch_bounds__(256) void flash_attn(const u16* __restrict__ Qp,
                                                  const u16* __restrict__ Kp,
                                                  const u16* __restrict__ VT,
                                                  const float* __restrict__ mask,
                                                  u16* __restrict__ Og) {
  constexpr int LDK = 72;              // 64 + 8 pad (144B rows, 16B aligned)
  __shared__ u16 Kt[2][64 * LDK];      // [m][d]
  __shared__ u16 Vt[2][64 * LDK];      // [d][m]
  __shared__ u16 Ss[8 * 16 * LDK];     // per-wave S scratch [row][m]

  const int t = threadIdx.x;
  const int lane = t & 31, wave = t >> 5;
  const int l16 = lane & 15, half = lane >> 4;
  const int b = blockIdx.z, h = blockIdx.y;
  const size_t boff = (size_t)b * (NPATCH * DIMC) + h * 64;   // row-major Q'/K'/O
  const size_t voff = (size_t)(h * 64) * BN + (size_t)b * NPATCH;  // VT slice
  const int n0 = blockIdx.x * 128 + wave * 16;

  // q' A-frags held in registers for the whole m loop
  FragB16 qf[2];
  {
    const u16* qrow = Qp + boff + (size_t)(n0 + l16) * DIMC;
#pragma unroll
    for (int f = 0; f < 2; ++f) {
      const u16* p = qrow + f * 32 + 8 * half;
      qf[f].q[0] = *(const u32x4*)p;
      qf[f].q[1] = *(const u32x4*)(p + 16);
    }
  }

  f32x8 oacc[4];
#pragma unroll
  for (int i = 0; i < 4; ++i) oacc[i] = fz8();
  float rs[8] = {0.f, 0.f, 0.f, 0.f, 0.f, 0.f, 0.f, 0.f};

  u16* sw = Ss + wave * (16 * LDK);

  auto fillkv = [&](int mc, int buf) {
    // K' chunk -> Kt[m][d]
#pragma unroll
    for (int i = 0; i < 2; ++i) {
      int idx = t + i * 256;
      int m = idx >> 3, dc = (idx & 7) * 8;
      async_copy_b128(Kp + boff + (size_t)(mc + m) * DIMC + dc,
                      &Kt[buf][m * LDK + dc]);
    }
    // V chunk (already transposed in memory) -> Vt[d][m]
#pragma unroll
    for (int i = 0; i < 2; ++i) {
      int idx = t + i * 256;
      int d = idx >> 3, m8 = (idx & 7) * 8;
      async_copy_b128(VT + voff + (size_t)d * BN + mc + m8,
                      &Vt[buf][d * LDK + m8]);
    }
  };

  fillkv(0, 0);
  for (int ic = 0; ic < NPATCH / 64; ++ic) {
    const int cur = ic & 1;
    const int mc = ic * 64;
    wait_async0();
    __syncthreads();
    if (ic + 1 < NPATCH / 64) fillkv(mc + 64, cur ^ 1);

    // S = q' @ k'^T, then mask, rowsum, spill bf16 to per-wave scratch
#pragma unroll
    for (int mt = 0; mt < 4; ++mt) {
      f32x8 s = fz8();
#pragma unroll
      for (int f = 0; f < 2; ++f) {
        FragB16 bk;
        const u16* p = &Kt[cur][(mt * 16 + l16) * LDK + f * 32 + 8 * half];
        bk.q[0] = *(const u32x4*)p;
        bk.q[1] = *(const u32x4*)(p + 16);
        s = wmma_bf16(qf[f], bk, s);
      }
      const float* mrow =
          mask + (size_t)(n0 + 8 * half) * NPATCH + mc + mt * 16 + l16;
#pragma unroll
      for (int r = 0; r < 8; ++r) {
        float sv = s[r] * mrow[(size_t)r * NPATCH];
        rs[r] += sv;
        sw[(r + 8 * half) * LDK + mt * 16 + l16] = f2bf(sv);
      }
    }
    // per-wave LDS RAW: wait on DScnt (CDNA5 split counters)
    asm volatile("s_wait_dscnt 0" ::: "memory");

    // Re-read S in A-frag layout, O += S @ V
    FragB16 sf[2];
#pragma unroll
    for (int p2 = 0; p2 < 2; ++p2) {
      const u16* p = sw + l16 * LDK + p2 * 32 + 8 * half;
      sf[p2].q[0] = *(const u32x4*)p;
      sf[p2].q[1] = *(const u32x4*)(p + 16);
    }
#pragma unroll
    for (int nt = 0; nt < 4; ++nt) {
#pragma unroll
      for (int p2 = 0; p2 < 2; ++p2) {
        FragB16 bv;
        const u16* p = &Vt[cur][(nt * 16 + l16) * LDK + p2 * 32 + 8 * half];
        bv.q[0] = *(const u32x4*)p;
        bv.q[1] = *(const u32x4*)(p + 16);
        oacc[nt] = wmma_bf16(sf[p2], bv, oacc[nt]);
      }
    }
  }

  // rowsum across the 16 lanes holding one row (wave32 halves), z = 1/(s+eps)
#pragma unroll
  for (int r = 0; r < 8; ++r) {
    float v = rs[r];
    v += __shfl_xor(v, 1);
    v += __shfl_xor(v, 2);
    v += __shfl_xor(v, 4);
    v += __shfl_xor(v, 8);
    rs[r] = 1.0f / (v + 1e-6f);
  }
#pragma unroll
  for (int nt = 0; nt < 4; ++nt)
#pragma unroll
    for (int r = 0; r < 8; ++r)
      Og[boff + (size_t)(n0 + 8 * half + r) * DIMC + nt * 16 + l16] =
          f2bf(oacc[nt][r] * rs[r]);
}

// ---------------------------------------------------------------------------
extern "C" void kernel_launch(void* const* d_in, const int* in_sizes, int n_in,
                              void* d_out, int out_size, void* d_ws, size_t ws_size,
                              hipStream_t stream) {
  const float* x    = (const float*)d_in[0];  // [4,1024,768]
  const float* Wqkv = (const float*)d_in[1];  // [768,2304]
  const float* Wout = (const float*)d_in[2];  // [768,768]
  const float* bout = (const float*)d_in[3];  // [768]
  const float* mask = (const float*)d_in[4];  // [1024,1024]
  float* out = (float*)d_out;                 // [4,1024,768] f32

  char* ws = (char*)d_ws;
  const size_t SEG2 = (size_t)SEG * 2;
  u16* QT  = (u16*)ws;                        // QT,KT,VT transposed [768][4096] segs
  u16* Qp  = (u16*)(ws + 3 * SEG2);           // Q',K' row-major (2*SEG)
  u16* O   = (u16*)(ws + 5 * SEG2);           // attention output, row-major (SEG)
  u16* Mt  = (u16*)(ws + 6 * SEG2);                        // mask^T bf16
  u16* xb  = (u16*)(ws + 6 * SEG2 + 2097152);              // x bf16 (row-major)
  u16* WqT = (u16*)(ws + 6 * SEG2 + 2097152 + 6291456);    // Wqkv^T bf16 [2304][768]
  u16* WoT = (u16*)(ws + 6 * SEG2 + 2097152 + 9830400);    // Wout^T bf16 [768][768]

  // one-time conversions / transposes of static operands (hw v_cvt)
  cvt_f32_bf16<<<dim3(3072), 256, 0, stream>>>(x, xb, BN * DIMC);
  transpose_cvt<<<dim3(72, 24), 256, 0, stream>>>(Wqkv, WqT, 768, 2304);
  transpose_cvt<<<dim3(24, 24), 256, 0, stream>>>(Wout, WoT, 768, 768);
  transpose_cvt<<<dim3(32, 32), 256, 0, stream>>>(mask, Mt, 1024, 1024);

  // qkv GEMM + relu/eps, transposed outputs:  [4096,768] x [768,2304]
  wgemm128<0><<<dim3(18, 32, 1), 256, 0, stream>>>(xb, WqT, QT, nullptr, 768, 768, 768);
  // graph mix: q' = q + 0.1*M^T q (z = b*2 + {q,k}):  [1024,1024] x [1024,768] x8
  wgemm128<1><<<dim3(6, 8, 8), 256, 0, stream>>>(Mt, QT, Qp, nullptr, 1024, BN, 1024);
  // fused masked linear attention (V consumed directly from transposed VT)
  flash_attn<<<dim3(8, 12, 4), 256, 0, stream>>>(Qp, Qp + SEG, QT + 2ull * SEG, mask, O);
  // final projection + bias: [4096,768] x [768,768]
  wgemm128<2><<<dim3(6, 32, 1), 256, 0, stream>>>(O, WoT, out, bout, 768, 768, 768);
}
